// VAE_79388175499581
// MI455X (gfx1250) — matrix operand order repaired and checked by reading, not compiled, not run on hardware
//
#include <hip/hip_runtime.h>
#include <hip/hip_bf16.h>

// ---------------------------------------------------------------------------
// VAE forward ELBO on MI455X (gfx1250), wave32, bf16 WMMA 16x16x32.
// B=65536, D=784, H=512, L=32.
// ---------------------------------------------------------------------------

typedef __attribute__((ext_vector_type(16))) __bf16 v16bf;
typedef __attribute__((ext_vector_type(8)))  float  v8f;
typedef __attribute__((ext_vector_type(8)))  unsigned short us8;
typedef __attribute__((ext_vector_type(4)))  float  f4;

#define B_ROWS 65536
#define DDIM   784
#define HDIM   512
#define LDIM   32

// padded LDS row stride (ushorts): 40 -> 80 B -> 20 dwords; 20*n mod 64 is a
// permutation over 16 lanes => conflict-free ds_load_b128 fragment fetches.
#define AST 40
#define XS_SZ (128 * AST)   // one A-tile buffer (128 rows x 32 k, padded)
#define WT_SZ (64 * AST)    // one B-tile buffer (64 cols x 32 k, padded)

// native f32 -> bf16 (RTNE) via hardware convert (v_cvt_pk_bf16_f32)
static __device__ __forceinline__ unsigned short f2bf(float f) {
  union { __bf16 b; unsigned short s; } u;
  u.b = (__bf16)f;
  return u.s;
}

union Frag {
  us8            h[2];
  unsigned short s[16];
  __bf16         b[16];
  v16bf          v;
};

// A-matrix fragment (16x32 bf16) from LDS tile [row][AST].
// lanes 0-15 -> M=lane, K=0..15; lanes 16-31 -> M=lane-16, K=16..31.
static __device__ __forceinline__ v16bf lds_a_frag(const unsigned short* xs, int rowbase) {
  const int lane = threadIdx.x & 31;
  const int row  = rowbase + (lane & 15);
  const int kh   = lane >> 4;
  Frag f;
  f.h[0] = *(const us8*)(xs + row * AST + 16 * kh);
  f.h[1] = *(const us8*)(xs + row * AST + 16 * kh + 8);
  return f.v;
}

// B-matrix fragment (32x16 bf16) from LDS tile [n][AST].
// lanes 0-15 -> N=lane, K=0..15; lanes 16-31 -> N=lane-16, K=16..31.
static __device__ __forceinline__ v16bf lds_b_frag(const unsigned short* wt, int nbase) {
  const int lane = threadIdx.x & 31;
  const int n    = nbase + (lane & 15);
  const int kh   = lane >> 4;
  const us8* p = (const us8*)(wt + n * AST + 16 * kh);
  Frag f;
  f.h[0] = p[0];
  f.h[1] = p[1];
  return f.v;
}

static __device__ __forceinline__ v8f wmma_bf16(v16bf a, v16bf b, v8f c) {
  return __builtin_amdgcn_wmma_f32_16x16x32_bf16(false, a, false, b, (short)0, c,
                                                 false, false);
}

// ---------------------------------------------------------------------------
__global__ void vae_zero_out(float* out) {
  if (threadIdx.x == 0 && blockIdx.x == 0) out[0] = 0.0f;
}

// ---------------------------------------------------------------------------
// Kernel 1: h = tanh(x @ We1 + be1)  ->  bf16 workspace [B, H]
// Block 256 thr (8 waves). Tile 128(M) x 64(N), K steps of 32, double-buffered.
// ---------------------------------------------------------------------------
__global__ __launch_bounds__(256)
void vae_enc_kernel(const float* __restrict__ x, const float* __restrict__ We1,
                    const float* __restrict__ be1, unsigned short* __restrict__ h_ws) {
  __shared__ __align__(16) unsigned short xs[2 * XS_SZ];
  __shared__ __align__(16) unsigned short wt[2 * WT_SZ];
  const int tid  = threadIdx.x;
  const int lane = tid & 31;
  const int wave = tid >> 5;
  const int kh   = lane >> 4;
  const int rowbase = blockIdx.x * 128;
  const int colbase = blockIdx.y * 64;

  auto stage = [&](int k0, int buf) {
    // x tile: thread -> (row, 16-wide k chunk). Chunk validity is uniform
    // (784 is a multiple of 16) => unpredicated b128 loads.
    {
      const int row = tid >> 1;
      const int kb  = (tid & 1) * 16;
      Frag f;
      if (k0 + kb < DDIM) {
        const f4* src = (const f4*)(x + (size_t)(rowbase + row) * DDIM + k0 + kb);
        if (k0 + kb + 64 < DDIM) __builtin_prefetch(src + 16, 0, 1);
#pragma unroll
        for (int q = 0; q < 4; ++q) {
          const f4 v = src[q];
#pragma unroll
          for (int j = 0; j < 4; ++j) f.b[q * 4 + j] = (__bf16)v[j];
        }
      } else {
#pragma unroll
        for (int e = 0; e < 16; ++e) f.s[e] = 0;
      }
      us8* dst = (us8*)(xs + buf * XS_SZ + row * AST + kb);
      dst[0] = f.h[0];
      dst[1] = f.h[1];
    }
    // We1 tile transposed: thread -> (k row, 8 contiguous n). wt[n][k].
    {
      const int kk = tid >> 3;         // 0..31
      const int n0 = (tid & 7) * 8;    // 0..56
      float t[8];
      if (k0 + kk < DDIM) {
        const f4* src = (const f4*)(We1 + (size_t)(k0 + kk) * HDIM + colbase + n0);
        const f4 a = src[0], b = src[1];
#pragma unroll
        for (int j = 0; j < 4; ++j) { t[j] = a[j]; t[4 + j] = b[j]; }
      } else {
#pragma unroll
        for (int j = 0; j < 8; ++j) t[j] = 0.0f;
      }
      unsigned short* w = wt + buf * WT_SZ;
#pragma unroll
      for (int j = 0; j < 8; ++j) w[(n0 + j) * AST + kk] = f2bf(t[j]);
    }
  };

  v8f c[4] = {};
  const int NS = (DDIM + 31) / 32;  // 25
  stage(0, 0);
  int buf = 0;
  for (int s = 0; s < NS; ++s) {
    __syncthreads();
    if (s + 1 < NS) stage((s + 1) * 32, buf ^ 1);
    // issue all fragment loads, then fence the scheduler so WMMAs use staged waits
    const v16bf a = lds_a_frag(xs + buf * XS_SZ, wave * 16);
    v16bf bf[4];
#pragma unroll
    for (int nt = 0; nt < 4; ++nt) bf[nt] = lds_b_frag(wt + buf * WT_SZ, nt * 16);
    __builtin_amdgcn_sched_barrier(0);
#pragma unroll
    for (int nt = 0; nt < 4; ++nt) c[nt] = wmma_bf16(a, bf[nt], c[nt]);
    buf ^= 1;
  }

#pragma unroll
  for (int nt = 0; nt < 4; ++nt) {
    const int col = colbase + nt * 16 + (lane & 15);
    const float bias = be1[col];
#pragma unroll
    for (int r = 0; r < 8; ++r) {
      const int row = rowbase + wave * 16 + r + 8 * kh;
      h_ws[(size_t)row * HDIM + col] = f2bf(tanhf(c[nt][r] + bias));
    }
  }
}

// ---------------------------------------------------------------------------
// Kernel 2: [mu|sig] = h @ [Wmu|Wsig] (K=512), reparameterize + KL terms,
// then hd = tanh(z @ Wd1 + bd1) (single K=32 WMMA step). 128 rows/block.
// ---------------------------------------------------------------------------
__global__ __launch_bounds__(256)
void vae_latent_kernel(const unsigned short* __restrict__ h_ws,
                       const float* __restrict__ eps,
                       const float* __restrict__ Wmu, const float* __restrict__ bmu,
                       const float* __restrict__ Wsig, const float* __restrict__ bsig,
                       const float* __restrict__ Wd1, const float* __restrict__ bd1,
                       unsigned short* __restrict__ hd_ws, float* out) {
  __shared__ __align__(16) unsigned short hs[2 * XS_SZ];
  __shared__ __align__(16) unsigned short wt[2 * WT_SZ];
  __shared__ __align__(16) unsigned short zs[XS_SZ];
  __shared__ float red[256];
  const int tid  = threadIdx.x;
  const int lane = tid & 31;
  const int wave = tid >> 5;
  const int kh   = lane >> 4;
  const int rowbase = blockIdx.x * 128;

  auto stage = [&](int k0, int buf) {
    {
      const int row = tid >> 1;
      const int kb  = (tid & 1) * 16;
      const us8* src = (const us8*)(h_ws + (size_t)(rowbase + row) * HDIM + k0 + kb);
      us8* dst = (us8*)(hs + buf * XS_SZ + row * AST + kb);
      dst[0] = src[0];
      dst[1] = src[1];
    }
    {
      const int kk = tid >> 3;
      const int n0 = (tid & 7) * 8;
      const f4* src = (n0 < 32)
          ? (const f4*)(Wmu + (size_t)(k0 + kk) * LDIM + n0)
          : (const f4*)(Wsig + (size_t)(k0 + kk) * LDIM + (n0 - 32));
      const f4 a = src[0], b = src[1];
      unsigned short* w = wt + buf * WT_SZ;
#pragma unroll
      for (int j = 0; j < 4; ++j) {
        w[(n0 + j) * AST + kk]     = f2bf(a[j]);
        w[(n0 + 4 + j) * AST + kk] = f2bf(b[j]);
      }
    }
  };

  // ---- GEMM1 ----
  v8f c[4] = {};
  const int NS = HDIM / 32;  // 16
  stage(0, 0);
  int buf = 0;
  for (int s = 0; s < NS; ++s) {
    __syncthreads();
    if (s + 1 < NS) stage((s + 1) * 32, buf ^ 1);
    const v16bf a = lds_a_frag(hs + buf * XS_SZ, wave * 16);
    v16bf bf[4];
#pragma unroll
    for (int nt = 0; nt < 4; ++nt) bf[nt] = lds_b_frag(wt + buf * WT_SZ, nt * 16);
    __builtin_amdgcn_sched_barrier(0);
#pragma unroll
    for (int nt = 0; nt < 4; ++nt) c[nt] = wmma_bf16(a, bf[nt], c[nt]);
    buf ^= 1;
  }

  // ---- reparameterize + KL (log_p_z - log_q = 0.5*(eps^2 + log(sig) - z^2)) ----
  float kl = 0.0f;
#pragma unroll
  for (int t = 0; t < 2; ++t) {
    const int n = t * 16 + (lane & 15);
    const float bm = bmu[n];
    const float bs = bsig[n];
#pragma unroll
    for (int r = 0; r < 8; ++r) {
      const int mloc = wave * 16 + r + 8 * kh;
      const int row  = rowbase + mloc;
      const float mu = c[t][r] + bm;
      const float sg = expf(c[t + 2][r] + bs);
      const float ev = eps[(size_t)row * LDIM + n];
      const float z  = mu + sqrtf(sg) * ev;
      kl += 0.5f * (ev * ev + logf(sg) - z * z);
      zs[mloc * AST + n] = f2bf(z);  // A-fragment layout for GEMM2
    }
  }
  __syncthreads();

  // ---- GEMM2: hd = tanh(z @ Wd1 + bd1); wave owns cols [wave*64, +64) ----
  v16bf bfrag[4];
#pragma unroll
  for (int nt = 0; nt < 4; ++nt) {
    const int col = wave * 64 + nt * 16 + (lane & 15);
    Frag f;
#pragma unroll
    for (int e = 0; e < 16; ++e) {
      const int k = e + 16 * kh;
      f.s[e] = f2bf(Wd1[(size_t)k * HDIM + col]);
    }
    bfrag[nt] = f.v;
  }
  for (int mt = 0; mt < 8; ++mt) {
    const v16bf a = lds_a_frag(zs, mt * 16);
    v8f d[4] = {};
    __builtin_amdgcn_sched_barrier(0);
#pragma unroll
    for (int nt = 0; nt < 4; ++nt) d[nt] = wmma_bf16(a, bfrag[nt], d[nt]);
#pragma unroll
    for (int nt = 0; nt < 4; ++nt) {
      const int col  = wave * 64 + nt * 16 + (lane & 15);
      const float bias = bd1[col];
#pragma unroll
      for (int r = 0; r < 8; ++r) {
        const int row = rowbase + mt * 16 + r + 8 * kh;
        hd_ws[(size_t)row * HDIM + col] = f2bf(tanhf(d[nt][r] + bias));
      }
    }
  }

  red[tid] = kl;
  __syncthreads();
  for (int s = 128; s > 0; s >>= 1) {
    if (tid < s) red[tid] += red[tid + s];
    __syncthreads();
  }
  if (tid == 0) atomicAdd(out, red[0] * (1.0f / (float)B_ROWS));
}

// ---------------------------------------------------------------------------
// Kernel 3: logits = hd @ Wd2 + bd2, fused Bernoulli log-likelihood vs x.
// Tile 128(M) x 64(N) over N=784 (last tile masked), K=512, double-buffered.
// ---------------------------------------------------------------------------
__global__ __launch_bounds__(256)
void vae_dec_kernel(const unsigned short* __restrict__ hd_ws,
                    const float* __restrict__ Wd2, const float* __restrict__ bd2,
                    const float* __restrict__ x, float* out) {
  __shared__ __align__(16) unsigned short hs[2 * XS_SZ];
  __shared__ __align__(16) unsigned short wt[2 * WT_SZ];
  __shared__ float red[256];
  const int tid  = threadIdx.x;
  const int lane = tid & 31;
  const int wave = tid >> 5;
  const int kh   = lane >> 4;
  const int rowbase = blockIdx.x * 128;
  const int colbase = blockIdx.y * 64;

  auto stage = [&](int k0, int buf) {
    {
      const int row = tid >> 1;
      const int kb  = (tid & 1) * 16;
      const us8* src = (const us8*)(hd_ws + (size_t)(rowbase + row) * HDIM + k0 + kb);
      us8* dst = (us8*)(hs + buf * XS_SZ + row * AST + kb);
      dst[0] = src[0];
      dst[1] = src[1];
    }
    {
      // Wd2 tile: 8-wide n chunks are uniformly valid (784 % 8 == 0).
      const int kk = tid >> 3;
      const int n0 = (tid & 7) * 8;
      float t[8];
      if (colbase + n0 < DDIM) {
        const f4* src = (const f4*)(Wd2 + (size_t)(k0 + kk) * DDIM + colbase + n0);
        const f4 a = src[0], b = src[1];
#pragma unroll
        for (int j = 0; j < 4; ++j) { t[j] = a[j]; t[4 + j] = b[j]; }
      } else {
#pragma unroll
        for (int j = 0; j < 8; ++j) t[j] = 0.0f;
      }
      unsigned short* w = wt + buf * WT_SZ;
#pragma unroll
      for (int j = 0; j < 8; ++j) w[(n0 + j) * AST + kk] = f2bf(t[j]);
    }
  };

  v8f c[4] = {};
  const int NS = HDIM / 32;  // 16
  stage(0, 0);
  int buf = 0;
  for (int s = 0; s < NS; ++s) {
    __syncthreads();
    if (s + 1 < NS) stage((s + 1) * 32, buf ^ 1);
    const v16bf a = lds_a_frag(hs + buf * XS_SZ, wave * 16);
    v16bf bf[4];
#pragma unroll
    for (int nt = 0; nt < 4; ++nt) bf[nt] = lds_b_frag(wt + buf * WT_SZ, nt * 16);
    __builtin_amdgcn_sched_barrier(0);
#pragma unroll
    for (int nt = 0; nt < 4; ++nt) c[nt] = wmma_bf16(a, bf[nt], c[nt]);
    buf ^= 1;
  }

  // Bernoulli: log sigmoid(t) = -softplus(-t); log(1 - sigmoid(t)) = -softplus(t)
  float acc = 0.0f;
#pragma unroll
  for (int nt = 0; nt < 4; ++nt) {
    const int col = colbase + nt * 16 + (lane & 15);
    if (col < DDIM) {
      const float bias = bd2[col];
#pragma unroll
      for (int r = 0; r < 8; ++r) {
        const int row = rowbase + wave * 16 + r + 8 * kh;
        const float t  = c[nt][r] + bias;
        const float xv = x[(size_t)row * DDIM + col];
        const float e  = log1pf(expf(-fabsf(t)));
        const float sp_pos = fmaxf(t, 0.0f) + e;   // softplus(t)
        const float sp_neg = fmaxf(-t, 0.0f) + e;  // softplus(-t)
        acc += (xv == 1.0f) ? -sp_neg : -sp_pos;
      }
    }
  }

  red[tid] = acc;
  __syncthreads();
  for (int s = 128; s > 0; s >>= 1) {
    if (tid < s) red[tid] += red[tid + s];
    __syncthreads();
  }
  if (tid == 0) atomicAdd(out, red[0] * (1.0f / (float)B_ROWS));
}

// ---------------------------------------------------------------------------
extern "C" void kernel_launch(void* const* d_in, const int* in_sizes, int n_in,
                              void* d_out, int out_size, void* d_ws, size_t ws_size,
                              hipStream_t stream) {
  (void)in_sizes; (void)n_in; (void)out_size; (void)ws_size;
  const float* x    = (const float*)d_in[0];
  const float* eps  = (const float*)d_in[1];
  const float* We1  = (const float*)d_in[2];
  const float* be1  = (const float*)d_in[3];
  const float* Wmu  = (const float*)d_in[4];
  const float* bmu  = (const float*)d_in[5];
  const float* Wsig = (const float*)d_in[6];
  const float* bsig = (const float*)d_in[7];
  const float* Wd1  = (const float*)d_in[8];
  const float* bd1  = (const float*)d_in[9];
  const float* Wd2  = (const float*)d_in[10];
  const float* bd2  = (const float*)d_in[11];
  float* out = (float*)d_out;

  // workspace: h (bf16, 64 MB) | hd (bf16, 64 MB)
  unsigned short* h_ws  = (unsigned short*)d_ws;
  unsigned short* hd_ws = h_ws + (size_t)B_ROWS * HDIM;

  vae_zero_out<<<1, 32, 0, stream>>>(out);
  vae_enc_kernel<<<dim3(B_ROWS / 128, HDIM / 64), 256, 0, stream>>>(x, We1, be1, h_ws);
  vae_latent_kernel<<<dim3(B_ROWS / 128), 256, 0, stream>>>(h_ws, eps, Wmu, bmu, Wsig,
                                                            bsig, Wd1, bd1, hd_ws, out);
  vae_dec_kernel<<<dim3(B_ROWS / 128, (DDIM + 63) / 64), 256, 0, stream>>>(hd_ws, Wd2,
                                                                           bd2, x, out);
}
